// DotProductAttention_46961172414943
// MI455X (gfx1250) — matrix operand order repaired and checked by reading, not compiled, not run on hardware
//
#include <hip/hip_runtime.h>
#include <stdint.h>

typedef __attribute__((ext_vector_type(16))) __bf16 v16bf;
typedef __attribute__((ext_vector_type(8)))  __bf16 v8bf;
typedef __attribute__((ext_vector_type(4)))  __bf16 bf16x4;
typedef __attribute__((ext_vector_type(2)))  __bf16 bf16x2;
typedef __attribute__((ext_vector_type(8)))  float  v8f;
typedef __attribute__((ext_vector_type(4)))  float  f32x4;
typedef __attribute__((ext_vector_type(4)))  unsigned int u32x4;
typedef __attribute__((ext_vector_type(4)))  int  i32x4;
typedef __attribute__((ext_vector_type(8)))  int  i32x8;

#ifndef __has_builtin
#define __has_builtin(x) 0
#endif

#if __has_builtin(__builtin_amdgcn_tensor_load_to_lds)
#define HAVE_TDM 1
#else
#define HAVE_TDM 0
#endif

__device__ __forceinline__ void tdm_wait0() {
#if __has_builtin(__builtin_amdgcn_s_wait_tensorcnt)
  __builtin_amdgcn_s_wait_tensorcnt(0);
#else
  asm volatile("s_wait_tensorcnt 0x0" ::: "memory");
#endif
}

#if HAVE_TDM
// 2D TDM load: 64 rows x 64 fp32 elements, row stride 64 elements, into LDS.
// D# packing per CDNA5 ISA (groups 2/3 zero => 2D).
__device__ __forceinline__ void tdm_load_tile(unsigned ldsOff, const void* gptr) {
  unsigned long long ga = (unsigned long long)(uintptr_t)gptr;
  u32x4 g0;
  g0.x = 1u;                                   // count=1, no gather
  g0.y = ldsOff;                               // lds_addr [63:32]
  g0.z = (unsigned)ga;                         // global_addr lo
  g0.w = (unsigned)((ga >> 32) & 0x01FFFFFFull) | (2u << 30); // addr[56:32] | type=2
  i32x8 g1;
  g1[0] = (int)(2u << 16);        // workgroup_mask=0, data_size=2 (4B)
  g1[1] = (int)(64u << 16);       // tensor_dim0 lo16 = 64 (bits 63:48)
  g1[2] = (int)(4096u << 16);     // dim0 hi16=0 | tensor_dim1 lo16 = 4096
  g1[3] = (int)(64u << 16);       // dim1 hi16=0 | tile_dim0 = 64
  g1[4] = (int)64;                // tile_dim1 = 64, tile_dim2 = 0
  g1[5] = 64;                     // tensor_dim0_stride lo32 = 64
  g1[6] = 0;                      // stride0 hi16 | stride1 lo16
  g1[7] = 0;                      // stride1 hi32
  i32x4 gz = {0, 0, 0, 0};
#if defined(__clang_major__) && (__clang_major__ >= 23)
  i32x8 gz8 = {0, 0, 0, 0, 0, 0, 0, 0};
  __builtin_amdgcn_tensor_load_to_lds(g0, g1, gz, gz, gz8, 0);
#else
  __builtin_amdgcn_tensor_load_to_lds(g0, g1, gz, gz, 0);
#endif
}
#endif

#define THREADS 128

__global__ __launch_bounds__(THREADS) void
fa_kernel(const float* __restrict__ Q, const float* __restrict__ V,
          const float* __restrict__ K, const int* __restrict__ mask,
          const int* __restrict__ scale_p, float* __restrict__ out)
{
  constexpr int S = 4096, D = 64;
  __shared__ __align__(16) float  lK32[2][64 * 64];   // fp32 TDM staging (K); [0] reused as O scratch
  __shared__ __align__(16) float  lV32[2][64 * 64];   // fp32 TDM staging (V)
  __shared__ __align__(16) __bf16 lKb[64 * 64];       // bf16 K, row-major [k][d]
  __shared__ __align__(16) __bf16 lVt[64 * 64];       // bf16 V, transposed [d][k]
  __shared__ __align__(16) __bf16 lP[4][16 * 64];     // per-wave P scratch

  const int b     = blockIdx.y;
  const int qtile = blockIdx.x;
  const int tid   = threadIdx.x;
  const int wave  = tid >> 5;
  const int lane  = tid & 31;
  const int half  = lane >> 4;
  const int l16   = lane & 15;

  const float rs = rsqrtf((float)scale_p[0]);   // 1/sqrt(64) = 0.125

  const float* Qb = Q + (size_t)b * S * D;
  const float* Kb = K + (size_t)b * S * D;
  const float* Vb = V + (size_t)b * S * D;
  const int*   mb = mask + (size_t)b * S;

  // ---- Q A-fragments (16x32 bf16 per K-chunk); scale folded in ----
  const int qrow = qtile * 64 + wave * 16 + l16;
  v16bf aQ[2];
#pragma unroll
  for (int c = 0; c < 2; ++c) {
    const float* qp = Qb + (size_t)qrow * D + c * 32 + half * 8;
    f32x4 q0 = *(const f32x4*)(qp);
    f32x4 q1 = *(const f32x4*)(qp + 4);
    f32x4 q2 = *(const f32x4*)(qp + 16);
    f32x4 q3 = *(const f32x4*)(qp + 20);
#pragma unroll
    for (int e = 0; e < 4; ++e) {
      aQ[c][e]      = (__bf16)(q0[e] * rs);
      aQ[c][4 + e]  = (__bf16)(q1[e] * rs);
      aQ[c][8 + e]  = (__bf16)(q2[e] * rs);
      aQ[c][12 + e] = (__bf16)(q3[e] * rs);
    }
  }

  v8f Ot[4];
  float m_i[8], l_i[8];
#pragma unroll
  for (int t = 0; t < 4; ++t)
#pragma unroll
    for (int r = 0; r < 8; ++r) Ot[t][r] = 0.f;
#pragma unroll
  for (int r = 0; r < 8; ++r) { m_i[r] = -3.0e38f; l_i[r] = 0.f; }

  // ---- prologue: stage key tile 0 into fp32 buffers ----
#if HAVE_TDM
  if (wave == 0) {
    tdm_load_tile((unsigned)(uintptr_t)&lK32[0][0], Kb);
    tdm_load_tile((unsigned)(uintptr_t)&lV32[0][0], Vb);
    tdm_wait0();
  }
#else
#pragma unroll
  for (int i = 0; i < 8; ++i) {
    int idx = (tid + i * THREADS) * 4;
    *(f32x4*)&lK32[0][idx] = *(const f32x4*)&Kb[idx];
    *(f32x4*)&lV32[0][idx] = *(const f32x4*)&Vb[idx];
  }
#endif
  __syncthreads();

  const int NKT = S / 64;
  for (int j = 0; j < NKT; ++j) {
    const int buf = j & 1;
#if HAVE_TDM
    if (wave == 0 && j + 1 < NKT) {   // async prefetch of next fp32 tile
      tdm_load_tile((unsigned)(uintptr_t)&lK32[buf ^ 1][0], Kb + (size_t)(j + 1) * 64 * D);
      tdm_load_tile((unsigned)(uintptr_t)&lV32[buf ^ 1][0], Vb + (size_t)(j + 1) * 64 * D);
    }
#endif
    // key-axis mask bias, issued early so the loads overlap the convert phase
    // (query-axis mask adds a per-row constant and cancels in softmax)
    int mraw[4];
#pragma unroll
    for (int t = 0; t < 4; ++t)
      mraw[t] = mb[j * 64 + t * 16 + l16];

    // ---- cooperative tile conversion: fp32 -> bf16, once per workgroup ----
    // K: row-major copy-convert (4 elems per step, b128 in / b64 out)
#pragma unroll
    for (int i = 0; i < 8; ++i) {
      int e = (tid + i * THREADS) * 4;
      f32x4 x = *(const f32x4*)&lK32[buf][e];
      bf16x4 y;
      y[0] = (__bf16)x[0]; y[1] = (__bf16)x[1];
      y[2] = (__bf16)x[2]; y[3] = (__bf16)x[3];
      *(bf16x4*)&lKb[e] = y;
    }
    // V: transpose-convert -> lVt[d][k] so P*V B-operands are contiguous
#pragma unroll
    for (int i = 0; i < 16; ++i) {
      int p = tid + i * THREADS;        // 0..2047: 64 d-cols x 32 k-pairs
      int d = p & 63;
      int k = (p >> 6) * 2;
      float a0 = lV32[buf][k * 64 + d];
      float a1 = lV32[buf][(k + 1) * 64 + d];
      bf16x2 y;
      y[0] = (__bf16)a0; y[1] = (__bf16)a1;
      *(bf16x2*)&lVt[d * 64 + k] = y;
    }
    __syncthreads();   // bf16 tiles ready

    float bias[4];
#pragma unroll
    for (int t = 0; t < 4; ++t)
      bias[t] = mraw[t] ? 0.f : -1.0e9f;

    // ---- S = Q K^T (+bias via C operand) ----
    v8f acc[4];
#pragma unroll
    for (int t = 0; t < 4; ++t) {
      v8f a;
#pragma unroll
      for (int e = 0; e < 8; ++e) a[e] = bias[t];
#pragma unroll
      for (int c = 0; c < 2; ++c) {
        const __bf16* kp = &lKb[(t * 16 + l16) * 64 + c * 32 + half * 8];
        v8bf b0 = *(const v8bf*)(kp);
        v8bf b1 = *(const v8bf*)(kp + 16);
        v16bf bK;
#pragma unroll
        for (int e = 0; e < 8; ++e) { bK[e] = b0[e]; bK[8 + e] = b1[e]; }
        a = __builtin_amdgcn_wmma_f32_16x16x32_bf16(false, aQ[c], false, bK,
                                                    (short)0, a, false, false);
      }
      acc[t] = a;
    }

    // ---- online softmax: row max / rescale / exp / row sum ----
#pragma unroll
    for (int r = 0; r < 8; ++r) {
      float v = fmaxf(fmaxf(acc[0][r], acc[1][r]), fmaxf(acc[2][r], acc[3][r]));
      v = fmaxf(v, __shfl_xor(v, 1));
      v = fmaxf(v, __shfl_xor(v, 2));
      v = fmaxf(v, __shfl_xor(v, 4));
      v = fmaxf(v, __shfl_xor(v, 8));
      float mn = fmaxf(m_i[r], v);
      float al = __expf(m_i[r] - mn);
      m_i[r] = mn;
      l_i[r] *= al;
#pragma unroll
      for (int t = 0; t < 4; ++t) Ot[t][r] *= al;
    }
#pragma unroll
    for (int t = 0; t < 4; ++t)
#pragma unroll
      for (int r = 0; r < 8; ++r)
        acc[t][r] = __expf(acc[t][r] - m_i[r]);
#pragma unroll
    for (int r = 0; r < 8; ++r) {
      float s = (acc[0][r] + acc[1][r]) + (acc[2][r] + acc[3][r]);
      s += __shfl_xor(s, 1);
      s += __shfl_xor(s, 2);
      s += __shfl_xor(s, 4);
      s += __shfl_xor(s, 8);
      l_i[r] += s;
    }

    // ---- P: C-layout -> A-layout via per-wave LDS scratch (bf16) ----
    __bf16* pw = &lP[wave][0];
#pragma unroll
    for (int t = 0; t < 4; ++t)
#pragma unroll
      for (int r = 0; r < 8; ++r)
        pw[(r + half * 8) * 64 + t * 16 + l16] = (__bf16)acc[t][r];

    v16bf aP[2];
#pragma unroll
    for (int c = 0; c < 2; ++c) {
      const __bf16* pp = pw + l16 * 64 + c * 32 + half * 8;
      v8bf p0 = *(const v8bf*)(pp);
      v8bf p1 = *(const v8bf*)(pp + 16);
#pragma unroll
      for (int e = 0; e < 8; ++e) { aP[c][e] = p0[e]; aP[c][8 + e] = p1[e]; }
    }

    // ---- O += P V  (B operand contiguous from transposed V) ----
#pragma unroll
    for (int t = 0; t < 4; ++t) {
      v8f o = Ot[t];
#pragma unroll
      for (int c = 0; c < 2; ++c) {
        const __bf16* vp = &lVt[(t * 16 + l16) * 64 + c * 32 + half * 8];
        v8bf b0 = *(const v8bf*)(vp);
        v8bf b1 = *(const v8bf*)(vp + 16);
        v16bf bV;
#pragma unroll
        for (int e = 0; e < 8; ++e) { bV[e] = b0[e]; bV[8 + e] = b1[e]; }
        o = __builtin_amdgcn_wmma_f32_16x16x32_bf16(false, aP[c], false, bV,
                                                    (short)0, o, false, false);
      }
      Ot[t] = o;
    }

#if HAVE_TDM
    if (wave == 0 && j + 1 < NKT) tdm_wait0();
    __syncthreads();   // compute done; next TDM tile visible; bf16 tiles free
#else
    __syncthreads();
    if (j + 1 < NKT) {
      const float* nk = Kb + (size_t)(j + 1) * 64 * D;
      const float* nv = Vb + (size_t)(j + 1) * 64 * D;
#pragma unroll
      for (int i = 0; i < 8; ++i) {
        int idx = (tid + i * THREADS) * 4;
        *(f32x4*)&lK32[buf ^ 1][idx] = *(const f32x4*)&nk[idx];
        *(f32x4*)&lV32[buf ^ 1][idx] = *(const f32x4*)&nv[idx];
      }
      __syncthreads();
    }
#endif
  }

  // ---- epilogue: normalize, stage O tile in LDS, store coalesced b128 ----
  float* lO = &lK32[0][0];   // fp32 staging buffer is free after the loop
#pragma unroll
  for (int r = 0; r < 8; ++r) {
    float inv = 1.0f / l_i[r];
    int row = wave * 16 + r + half * 8;      // row within the 64-row q tile
#pragma unroll
    for (int t = 0; t < 4; ++t)
      lO[row * 64 + t * 16 + l16] = Ot[t][r] * inv;
  }
  __syncthreads();
  {
    float* ob = out + ((size_t)b * S + (size_t)qtile * 64) * D;  // contiguous 64x64 tile
#pragma unroll
    for (int i = 0; i < 8; ++i) {
      int idx = (tid + i * THREADS) * 4;
      *(f32x4*)&ob[idx] = *(const f32x4*)&lO[idx];
    }
  }
}

extern "C" void kernel_launch(void* const* d_in, const int* in_sizes, int n_in,
                              void* d_out, int out_size, void* d_ws, size_t ws_size,
                              hipStream_t stream) {
  // setup_inputs order: query, value, key, attention_mask, scale_factor
  const float* Q    = (const float*)d_in[0];
  const float* V    = (const float*)d_in[1];
  const float* K    = (const float*)d_in[2];
  const int*   mask = (const int*)d_in[3];
  const int*   scl  = (const int*)d_in[4];
  float* out = (float*)d_out;
  dim3 grid(4096 / 64, 4, 1);
  fa_kernel<<<grid, THREADS, 0, stream>>>(Q, V, K, mask, scl, out);
  (void)in_sizes; (void)n_in; (void)out_size; (void)d_ws; (void)ws_size;
}
